// DGCNN_43456479101191
// MI455X (gfx1250) — matrix operand order repaired and checked by reading, not compiled, not run on hardware
//
#include <hip/hip_runtime.h>
#include <hip/hip_bf16.h>

typedef __attribute__((ext_vector_type(16))) _Float16 v16h;
typedef __attribute__((ext_vector_type(8)))  _Float16 v8h;
typedef __attribute__((ext_vector_type(4)))  _Float16 v4h;
typedef __attribute__((ext_vector_type(8)))  float    v8f;
typedef __attribute__((ext_vector_type(4)))  float    v4f;

#define NPG   100          // nodes per graph
#define MPAD  112          // 7 WMMA M-tiles of 16
#define EPG   1600         // edges per graph (E/B)
#define NBLK  2000         // graphs
#define NEDGE 3200000
#define HID   32
#define LATD  97
#define KTOP  60
#define FIN   128

#if __has_builtin(__builtin_amdgcn_tanhf)
__device__ __forceinline__ float fast_tanh(float v) { return __builtin_amdgcn_tanhf(v); }
#else
__device__ __forceinline__ float fast_tanh(float v) { return tanhf(v); }
#endif

// One workgroup per graph, fully fused in LDS. GCN matmuls on
// v_wmma_f32_16x16x32_f16; edge aggregation as a packed-CSR gather
// (b64 metadata + b128 data per edge), tanh on the gfx1250 v_tanh trans op.
__global__ __launch_bounds__(256) void dgcnn_fused(
    const float* __restrict__ x,  const int* __restrict__ ei,
    const float* __restrict__ W0, const float* __restrict__ b0,
    const float* __restrict__ W1, const float* __restrict__ b1,
    const float* __restrict__ W2, const float* __restrict__ b2,
    const float* __restrict__ W3, const float* __restrict__ b3,
    const float* __restrict__ c1w, const float* __restrict__ c1b,
    const float* __restrict__ c2w, const float* __restrict__ c2b,
    const float* __restrict__ l1w, const float* __restrict__ l1b,
    const float* __restrict__ l2w, const float* __restrict__ l2b,
    float* __restrict__ out)
{
  __shared__ __align__(32) _Float16 sH[MPAD][FIN];  // activations f16; later topk (f32)
  __shared__ __align__(32) _Float16 sWt[HID][FIN];  // W^T for WMMA B; later conv1 weights
  __shared__ __align__(16) float sHW[MPAD * HID];   // h@W (padded rows); later conv1 out
  __shared__ float sXC[NPG][LATD];                  // layer-output concat; later conv2 weights
  __shared__ unsigned short sSrc[EPG];              // raw edges (build-time only)
  __shared__ unsigned short sDst[EPG];
  __shared__ __align__(8) int2 sCsr[EPG];           // packed (src, norm) by dst; later scratch
  __shared__ int   sOff[NPG + 1];
  __shared__ int   sDeg[NPG];                       // degree, then CSR cursor
  __shared__ float sDinv[NPG];
  __shared__ float sW3[HID];
  __shared__ float sBias[HID];
  __shared__ int   sSel[KTOP];

  const int g    = blockIdx.x;
  const int tid  = threadIdx.x;
  const int lane = tid & 31;
  const int wv   = tid >> 5;
  const int base = g * NPG;

  __builtin_prefetch(&x[(size_t)base * FIN + (size_t)tid * 16], 0, 0);

  // zero M-padding rows (stay zero for every layer's A fragments)
  for (int i = tid; i < (MPAD - NPG) * FIN; i += 256)
    sH[NPG + i / FIN][i % FIN] = (_Float16)0.f;
  for (int i = tid; i < NPG; i += 256) sDeg[i] = 1;  // self-loop

  // x tile -> f16 LDS, b128 global loads / b64 LDS stores
  {
    const v4f* xv = (const v4f*)(x + (size_t)base * FIN);
    for (int i = tid; i < NPG * FIN / 4; i += 256) {
      const v4f v = xv[i];
      v4h h; h[0] = (_Float16)v[0]; h[1] = (_Float16)v[1];
             h[2] = (_Float16)v[2]; h[3] = (_Float16)v[3];
      const int idx4 = i * 4;
      *(v4h*)&sH[idx4 >> 7][idx4 & 127] = h;
    }
  }
  const int* srcg = ei;
  const int* dstg = ei + NEDGE;
  for (int e = tid; e < EPG; e += 256) {
    sSrc[e] = (unsigned short)(srcg[g * EPG + e] - base);
    sDst[e] = (unsigned short)(dstg[g * EPG + e] - base);
  }
  __syncthreads();

  for (int e = tid; e < EPG; e += 256) atomicAdd(&sDeg[sDst[e]], 1);
  __syncthreads();
  for (int i = tid; i < NPG; i += 256) sDinv[i] = rsqrtf((float)sDeg[i]);
  if (tid == 0) {  // exclusive scan of edge-only in-degree (100 elems)
    int acc = 0;
    for (int i = 0; i < NPG; ++i) { sOff[i] = acc; acc += sDeg[i] - 1; }
    sOff[NPG] = acc;
  }
  __syncthreads();
  for (int i = tid; i < NPG; i += 256) sDeg[i] = 0;  // CSR cursor
  __syncthreads();
  for (int e = tid; e < EPG; e += 256) {  // counting-sort by dst, packed records
    const int s = sSrc[e], d = sDst[e];
    const int p = sOff[d] + atomicAdd(&sDeg[d], 1);
    sCsr[p] = make_int2(s, __float_as_int(sDinv[s] * sDinv[d]));
  }
  __syncthreads();

  const float* Ws[3] = {W0, W1, W2};
  const float* bs[3] = {b0, b1, b2};

  for (int l = 0; l < 3; ++l) {
    const int Kdim = (l == 0) ? FIN : HID;
    for (int i = tid; i < Kdim * HID; i += 256)
      sWt[i % HID][i / HID] = (_Float16)Ws[l][i];
    for (int i = tid; i < HID; i += 256) sBias[i] = bs[l][i];
    __syncthreads();

    // ---- WMMA: sHW[112,32] = sH[112,Kdim] @ W[Kdim,32]; 7x2 tiles / 8 waves ----
    const int ksteps = Kdim / 32;
    for (int t = wv; t < 14; t += 8) {
      const int mt = t >> 1, nt = t & 1;
      const int r  = mt * 16 + (lane & 15);      // A row
      const int kh = (lane >> 4) * 16;           // A K base (16 contiguous halves/lane)
      const int n  = nt * 16 + (lane & 15);      // B col
      const int kb = (lane >> 4) * 16;           // B K base (16 contiguous halves/lane)
      v8f c = {};
      for (int kk = 0; kk < ksteps; ++kk) {
        v8h a0 = *(const v8h*)&sH[r][kk * 32 + kh];
        v8h a1 = *(const v8h*)&sH[r][kk * 32 + kh + 8];
        v16h a;
        #pragma unroll
        for (int e2 = 0; e2 < 8; ++e2) { a[e2] = a0[e2]; a[e2 + 8] = a1[e2]; }
        v16h bf = *(const v16h*)&sWt[n][kk * 32 + kb];
        c = __builtin_amdgcn_wmma_f32_16x16x32_f16(false, a, false, bf,
                                                   (short)0, c, false, false);
      }
      const int mb = mt * 16 + ((lane >> 4) << 3);
      const int nn = nt * 16 + (lane & 15);
      #pragma unroll
      for (int v = 0; v < 8; ++v) sHW[(mb + v) * HID + nn] = c[v];
    }
    __syncthreads();

    // ---- packed-CSR gather + bias + tanh: 100 nodes x 8 channel-quads ----
    for (int task = tid; task < NPG * 8; task += 256) {
      const int node = task >> 3;
      const int cg   = (task & 7) << 2;
      const float di = sDinv[node];
      v4f acc = (di * di) * *(const v4f*)&sHW[node * HID + cg];  // self-loop
      const int p1 = sOff[node + 1];
      for (int p = sOff[node]; p < p1; ++p) {
        const int2 r = sCsr[p];                                   // ds_load_b64
        acc += __int_as_float(r.y) * *(const v4f*)&sHW[r.x * HID + cg];
      }
      v4h hq;
      #pragma unroll
      for (int j = 0; j < 4; ++j) {
        const float v = fast_tanh(acc[j] + sBias[cg + j]);        // v_tanh_f32
        hq[j] = (_Float16)v;
        sXC[node][l * HID + cg + j] = v;
      }
      *(v4h*)&sH[node][cg] = hq;
    }
    __syncthreads();
  }

  // ---- layer 3: matvec [100,32]@[32,1], CSR gather, -> xc[:,96] ----
  for (int i = tid; i < HID; i += 256) sW3[i] = W3[i];
  __syncthreads();
  for (int i = tid; i < NPG; i += 256) {
    float s = 0.f;
    #pragma unroll
    for (int c2 = 0; c2 < HID; ++c2) s += (float)sH[i][c2] * sW3[c2];
    sHW[i] = s;
  }
  __syncthreads();
  const float bias3 = b3[0];
  for (int i = tid; i < NPG; i += 256) {
    const float di = sDinv[i];
    float acc = di * di * sHW[i];
    const int p1 = sOff[i + 1];
    for (int p = sOff[i]; p < p1; ++p) {
      const int2 r = sCsr[p];
      acc += __int_as_float(r.y) * sHW[r.x];
    }
    sXC[i][96] = fast_tanh(acc + bias3);
  }
  __syncthreads();

  // ---- global_sort_pool: stable descending rank, top-60 ----
  if (tid < NPG) {
    const float v = sXC[tid][96];
    int rank = 0;
    for (int j = 0; j < NPG; ++j) {
      const float vj = sXC[j][96];
      rank += (vj > v) || (vj == v && j < tid);
    }
    if (rank < KTOP) sSel[rank] = tid;
  }
  __syncthreads();

  // phase A: top-k copy (reads sXC) + stage conv1 weights into dead sWt region
  float* sTopk = reinterpret_cast<float*>(&sH[0][0]);   // [KTOP][LATD]
  float* sC1W  = reinterpret_cast<float*>(&sWt[0][0]);  // 1552 w + 16 b <= 2048 floats
  for (int i = tid; i < KTOP * LATD; i += 256)
    sTopk[i] = sXC[sSel[i / LATD]][i % LATD];
  for (int i = tid; i < 16 * LATD + 16; i += 256)
    sC1W[i] = (i < 16 * LATD) ? c1w[i] : c1b[i - 16 * LATD];
  __syncthreads();

  // phase B: conv1 [16][60] relu + stage conv2 weights into dead sXC region
  float* sOut1 = sHW;
  float* sC2W  = reinterpret_cast<float*>(&sXC[0][0]);  // 2560 w + 32 b
  for (int i = tid; i < 16 * KTOP; i += 256) {
    const int c2 = i / KTOP, k = i % KTOP;
    float s = sC1W[16 * LATD + c2];
    const float* w = &sC1W[c2 * LATD];
    const float* z = &sTopk[k * LATD];
    for (int d = 0; d < LATD; ++d) s += z[d] * w[d];
    sOut1[c2 * KTOP + k] = fmaxf(s, 0.f);
  }
  for (int i = tid; i < 32 * 80 + 32; i += 256)
    sC2W[i] = (i < 32 * 80) ? c2w[i] : c2b[i - 32 * 80];
  __syncthreads();

  // tail scratch overlays the (now dead) CSR buffer: 3200 floats
  float* sScr   = reinterpret_cast<float*>(&sCsr[0]);
  float* sPoolM = sScr;                 // [16*30]
  float* sFl    = sScr + 512;           // [832]
  float* sO1    = sScr + 1344;          // [128]
  float* sPart  = sScr + 1536;          // [256]
  for (int i = tid; i < 16 * 30; i += 256) {  // maxpool2
    const int c2 = i / 30, t = i % 30;
    sPoolM[i] = fmaxf(sOut1[c2 * KTOP + 2 * t], sOut1[c2 * KTOP + 2 * t + 1]);
  }
  __syncthreads();
  for (int i = tid; i < 32 * 26; i += 256) {  // conv2 k=5 -> fl[c*26+t]
    const int c2 = i / 26, t = i % 26;
    float s = sC2W[32 * 80 + c2];
    for (int ci = 0; ci < 16; ++ci) {
      const float* w = &sC2W[c2 * 80 + ci * 5];
      const float* p = &sPoolM[ci * 30 + t];
      #pragma unroll
      for (int j = 0; j < 5; ++j) s += p[j] * w[j];
    }
    sFl[i] = fmaxf(s, 0.f);
  }
  __syncthreads();
  {                                           // lin1: 832->128, split over all 8 waves
    const int j    = tid & 127;
    const int half = tid >> 7;                // 0: i in [0,416), 1: [416,832)
    float s = (half == 0) ? l1b[j] : 0.f;
    const int i0 = half * 416;
    for (int i2 = i0; i2 < i0 + 416; ++i2) s += sFl[i2] * l1w[i2 * 128 + j];
    sPart[tid] = s;
  }
  __syncthreads();
  if (tid < 128) sO1[tid] = fmaxf(sPart[tid] + sPart[tid + 128], 0.f);
  __syncthreads();
  if (tid == 0) {                             // lin2: 128->1
    float s = l2b[0];
    #pragma unroll
    for (int j = 0; j < 128; ++j) s += sO1[j] * l2w[j];
    out[g] = s;
  }
}

extern "C" void kernel_launch(void* const* d_in, const int* in_sizes, int n_in,
                              void* d_out, int out_size, void* d_ws, size_t ws_size,
                              hipStream_t stream) {
  const float* x   = (const float*)d_in[0];
  const int*   ei  = (const int*)d_in[1];
  const float* W0  = (const float*)d_in[3];  const float* b0 = (const float*)d_in[4];
  const float* W1  = (const float*)d_in[5];  const float* b1 = (const float*)d_in[6];
  const float* W2  = (const float*)d_in[7];  const float* b2 = (const float*)d_in[8];
  const float* W3  = (const float*)d_in[9];  const float* b3 = (const float*)d_in[10];
  const float* c1w = (const float*)d_in[11]; const float* c1b = (const float*)d_in[12];
  const float* c2w = (const float*)d_in[13]; const float* c2b = (const float*)d_in[14];
  const float* l1w = (const float*)d_in[15]; const float* l1b = (const float*)d_in[16];
  const float* l2w = (const float*)d_in[17]; const float* l2b = (const float*)d_in[18];
  float* out = (float*)d_out;

  dgcnn_fused<<<NBLK, 256, 0, stream>>>(x, ei, W0, b0, W1, b1, W2, b2, W3, b3,
                                        c1w, c1b, c2w, c2b, l1w, l1b, l2w, l2b, out);
}
